// IntentCapture_20358144983082
// MI455X (gfx1250) — matrix operand order repaired and checked by reading, not compiled, not run on hardware
//
#include <hip/hip_runtime.h>
#include <hip/hip_bf16.h>

// ---------------------------------------------------------------------------
// B=128, S=512, D=1024.  scores = sum_n tanh(X@W1 + b1)[.,n] * w2[n] fused into
// a bf16 WMMA GEMM (h never materialized).  Then masked softmax + attn^T X.
//
// Kernel 0: W1 fp32 -> W1T bf16 [n][k] (transposed) once per call.
// Kernel 1: per block: persist 128-row A tile as bf16 in LDS (258 KB, CDNA5
//           320KB LDS), stream B tiles via async global->LDS double buffer,
//           8 v_wmma per K-step per wave, hw-tanh epilogue.
// Kernel 2: masked softmax over S + u_att weighted sum + u_last copy.
// ---------------------------------------------------------------------------
#define DD      1024
#define SS      512
#define BB      128
#define ROWS    (BB * SS)
#define M_TILE  128
#define N_CHUNK 128
#define K_TILE  32
#define KSTEPS  (DD / K_TILE)
#define AROW    1032                  // padded A row (ushorts): 2064B = 516 banks ≡ 4 mod 64
#define BROW    40                    // padded B row (ushorts): 80B   = 20 banks
#define SMEM_USHORTS (M_TILE * AROW + 2 * N_CHUNK * BROW)
#define SMEM_BYTES   (SMEM_USHORTS * 2)   // 284,672 B  (< 320KB WGP LDS)

typedef __bf16 v16bf __attribute__((ext_vector_type(16)));
typedef __bf16 v2bf  __attribute__((ext_vector_type(2)));
typedef float  v8f   __attribute__((ext_vector_type(8)));
typedef int    v4i   __attribute__((ext_vector_type(4)));

union Frag16 { uint4 u[2]; v16bf v; };

// ---- fp32 -> bf16 (RNE), packed pair --------------------------------------
__device__ __forceinline__ unsigned short f2bf(float f) {
    unsigned u = __builtin_bit_cast(unsigned, f);
    u += 0x7FFFu + ((u >> 16) & 1u);
    return (unsigned short)(u >> 16);
}
__device__ __forceinline__ unsigned pk_bf16(float a, float b) {
#if __has_builtin(__builtin_amdgcn_cvt_pk_bf16_f32)
    v2bf r = __builtin_amdgcn_cvt_pk_bf16_f32(a, b);
    return __builtin_bit_cast(unsigned short2, r).x |
           ((unsigned)__builtin_bit_cast(unsigned short2, r).y << 16);
#else
    return (unsigned)f2bf(a) | ((unsigned)f2bf(b) << 16);
#endif
}

// ---- hardware tanh (gfx1250 V_TANH_F32), branchless -----------------------
__device__ __forceinline__ float htanh(float x) {
#if __has_builtin(__builtin_amdgcn_tanhf)
    return __builtin_amdgcn_tanhf(x);
#else
    float r;
    asm volatile("v_tanh_f32 %0, %1\n\tv_nop\n\tv_nop" : "=v"(r) : "v"(x));
    return r;
#endif
}

// ---- async global -> LDS (CDNA5), with synchronous fallback ---------------
#if __has_builtin(__builtin_amdgcn_global_load_async_to_lds_b128) && \
    __has_builtin(__builtin_amdgcn_s_wait_asynccnt)
#define HAVE_ASYNC_LDS 1
#else
#define HAVE_ASYNC_LDS 0
#endif

#if HAVE_ASYNC_LDS
// builtin signature (from hipcc diagnostic): param0 = <4 x i32>* addrspace(1),
// param1 = <4 x i32>* addrspace(3), then two immediate ints (offset, cpol).
typedef __attribute__((address_space(1))) v4i* g_v4i_p;
typedef __attribute__((address_space(3))) v4i* l_v4i_p;
__device__ __forceinline__ void async_b128(unsigned short* lds, const unsigned short* g) {
    __builtin_amdgcn_global_load_async_to_lds_b128(
        (g_v4i_p)g, (l_v4i_p)lds, 0, 0);
}
#endif

// ===========================================================================
// Kernel 0: W1T[n][k] = bf16(W1[k][n])   (1024x1024), 1024 blocks x 256 thr
// ===========================================================================
__global__ __launch_bounds__(256) void transpose_w1_kernel(
    const float* __restrict__ W1, unsigned short* __restrict__ w1t)
{
    __shared__ float tile[32][33];
    const int tid = threadIdx.x;
    const int bx = blockIdx.x & 31;        // n tile
    const int by = blockIdx.x >> 5;        // k tile
    const int r  = tid >> 3;               // 0..31
    const int c0 = (tid & 7) * 4;

    float4 v = *(const float4*)(W1 + (size_t)(by * 32 + r) * DD + bx * 32 + c0);
    tile[r][c0 + 0] = v.x; tile[r][c0 + 1] = v.y;
    tile[r][c0 + 2] = v.z; tile[r][c0 + 3] = v.w;
    __syncthreads();

    // write row n = bx*32+r, cols k = by*32 + c0..c0+3  (value = tile[k][n])
    unsigned p0 = pk_bf16(tile[c0 + 0][r], tile[c0 + 1][r]);
    unsigned p1 = pk_bf16(tile[c0 + 2][r], tile[c0 + 3][r]);
    unsigned* dst = (unsigned*)(w1t + (size_t)(bx * 32 + r) * DD + by * 32 + c0);
    dst[0] = p0; dst[1] = p1;
}

// ===========================================================================
// Kernel 1: fused GEMM + tanh + dot(w2) -> scores.  512 blocks x 256 thr.
// Wave w exclusively owns rows [blk*128 + w*16, +16): plain stores, no atomics.
// ===========================================================================
__global__ __launch_bounds__(256) void gemm_tanh_score_kernel(
    const float*          __restrict__ x,     // (ROWS, D) fp32
    const unsigned short* __restrict__ w1t,   // (D, D)  bf16, [n][k]
    const float*          __restrict__ b1,
    const float*          __restrict__ w2,
    float*                __restrict__ scores)
{
    extern __shared__ unsigned short smem[];
    unsigned short* a_full = smem;                       // [M_TILE][AROW]
    unsigned short* b_tile = smem + M_TILE * AROW;       // [2][N_CHUNK][BROW]

    const int tid  = threadIdx.x;
    const int wave = tid >> 5;
    const int lane = tid & 31;
    const int m0   = blockIdx.x * M_TILE;

    const int lrow = lane & 15;
    const int kA0  = (lane < 16) ? 0 : 8;    // A 16-bit 16x32 lane layout
    const int kB0  = (lane < 16) ? 0 : 8;    // B 16-bit 32x16 lane layout

    // ---- Phase 1: stage entire 128 x 1024 A tile once, fp32 -> bf16 -------
    for (int f = tid; f < M_TILE * (DD / 4); f += 256) {
        const int row = f >> 8;              // / (DD/4)
        const int c4  = f & 255;
        float4 v = *(const float4*)(x + (size_t)(m0 + row) * DD + c4 * 4);
        uint2 p; p.x = pk_bf16(v.x, v.y); p.y = pk_bf16(v.z, v.w);
        *(uint2*)(a_full + (size_t)row * AROW + c4 * 4) = p;
    }
    __syncthreads();

    const unsigned short* a_base = a_full + (size_t)(wave * 16 + lrow) * AROW;

    // per-thread B copy geometry: 32 bytes (one n row x 16 bf16 half) each
    const int bn   = tid >> 1;               // 0..127
    const int koff = (tid & 1) * 16;         // 0 / 16
    unsigned short* ldst0 = b_tile + (size_t)bn * BROW + koff;
    unsigned short* ldst1 = ldst0 + (size_t)N_CHUNK * BROW;   // buffer 1

    v8f sacc;
    #pragma unroll
    for (int e = 0; e < 8; ++e) sacc[e] = 0.0f;

    for (int n0 = 0; n0 < DD; n0 += N_CHUNK) {
        v8f acc[8];
        #pragma unroll
        for (int s = 0; s < 8; ++s)
            #pragma unroll
            for (int e = 0; e < 8; ++e) acc[s][e] = 0.0f;

        const unsigned short* gsrc_row = w1t + (size_t)(n0 + bn) * DD + koff;

        // ---- prologue: stage B k-tile 0 into buffer 0 ----------------------
#if HAVE_ASYNC_LDS
        async_b128(ldst0,      gsrc_row);
        async_b128(ldst0 + 8,  gsrc_row + 8);
        __builtin_amdgcn_s_wait_asynccnt(0);
#else
        ((uint4*)ldst0)[0] = ((const uint4*)gsrc_row)[0];
        ((uint4*)ldst0)[1] = ((const uint4*)gsrc_row)[1];
#endif
        __syncthreads();

        for (int ks = 0; ks < KSTEPS; ++ks) {
            const int cur = ks & 1;
            const bool more = (ks + 1) < KSTEPS;
            unsigned short* lnext = (cur ? ldst0 : ldst1);
            const unsigned short* gnext = gsrc_row + (ks + 1) * K_TILE;

            // ---- issue next B tile copy (overlaps the 8 WMMAs below) ------
#if HAVE_ASYNC_LDS
            if (more) { async_b128(lnext, gnext); async_b128(lnext + 8, gnext + 8); }
#else
            uint4 nb0, nb1;
            if (more) { nb0 = ((const uint4*)gnext)[0]; nb1 = ((const uint4*)gnext)[1]; }
#endif

            // ---- compute: A frag from persistent LDS, 8 B subtiles --------
            const int kk = ks * K_TILE;
            Frag16 af;
            af.u[0] = *(const uint4*)(a_base + kk + kA0);
            af.u[1] = *(const uint4*)(a_base + kk + kA0 + 16);

            const unsigned short* bbase =
                b_tile + (size_t)cur * N_CHUNK * BROW + (size_t)lrow * BROW + kB0;
            #pragma unroll
            for (int sub = 0; sub < 8; ++sub) {
                const uint4* bp = (const uint4*)(bbase + sub * 16 * BROW);
                Frag16 bf; bf.u[0] = bp[0]; bf.u[1] = bp[2];
                acc[sub] = __builtin_amdgcn_wmma_f32_16x16x32_bf16(
                    false, af.v, false, bf.v, (short)0, acc[sub], false, false);
            }

            // ---- retire the copy, flip buffers ----------------------------
#if HAVE_ASYNC_LDS
            __builtin_amdgcn_s_wait_asynccnt(0);
#else
            if (more) { ((uint4*)lnext)[0] = nb0; ((uint4*)lnext)[1] = nb1; }
#endif
            __syncthreads();
        }

        // ---- epilogue: tanh(+b1) and dot with w2, all in registers --------
        // C layout: elem r <-> (M = r or 8+r by lane half, N = lane&15)
        #pragma unroll
        for (int sub = 0; sub < 8; ++sub) {
            const int ng = n0 + sub * 16 + lrow;
            const float bv = b1[ng];
            const float wv = w2[ng];
            #pragma unroll
            for (int r = 0; r < 8; ++r)
                sacc[r] += htanh(acc[sub][r] + bv) * wv;
        }
    }

    // ---- reduce the 16 N-lanes of each half, store scores -----------------
    #pragma unroll
    for (int r = 0; r < 8; ++r) {
        float v = sacc[r];
        v += __shfl_xor(v, 1, 32);
        v += __shfl_xor(v, 2, 32);
        v += __shfl_xor(v, 4, 32);
        v += __shfl_xor(v, 8, 32);
        if ((lane & 15) == 0)
            scores[m0 + wave * 16 + ((lane >> 4) * 8) + r] = v;
    }
}

// ===========================================================================
// Kernel 2: masked softmax over S, u_att = attn^T X, u_last copy.  128 blocks.
// ===========================================================================
__global__ __launch_bounds__(256) void softmax_attend_kernel(
    const float* __restrict__ x,
    const int*   __restrict__ mask,
    const float* __restrict__ scores,
    float* __restrict__ out)             // [u_last (B*D) | u_att (B*D)]
{
    const int b = blockIdx.x;
    const int tid = threadIdx.x;
    const int lane = tid & 31, wave = tid >> 5;

    __shared__ float attn[SS];
    __shared__ float wmax[8];
    __shared__ float wsum[8];

    float v0 = scores[b * SS + tid];
    float v1 = scores[b * SS + 256 + tid];
    if (mask[b * SS + tid])       v0 = -1.0e9f;
    if (mask[b * SS + 256 + tid]) v1 = -1.0e9f;

    float m = fmaxf(v0, v1);
    for (int off = 16; off; off >>= 1) m = fmaxf(m, __shfl_xor(m, off, 32));
    if (lane == 0) wmax[wave] = m;
    __syncthreads();
    float bm = wmax[0];
    #pragma unroll
    for (int i = 1; i < 8; ++i) bm = fmaxf(bm, wmax[i]);

    float e0 = __expf(v0 - bm);
    float e1 = __expf(v1 - bm);
    float s = e0 + e1;
    for (int off = 16; off; off >>= 1) s += __shfl_xor(s, off, 32);
    if (lane == 0) wsum[wave] = s;
    __syncthreads();
    float bs = 0.0f;
    #pragma unroll
    for (int i = 0; i < 8; ++i) bs += wsum[i];
    const float inv = 1.0f / bs;

    attn[tid]       = e0 * inv;
    attn[tid + 256] = e1 * inv;
    __syncthreads();

    const float4* xb = (const float4*)(x + (size_t)b * SS * DD);
    float4 acc = make_float4(0.f, 0.f, 0.f, 0.f);
    for (int sidx = 0; sidx < SS; ++sidx) {
        const float a = attn[sidx];
        float4 xv = xb[sidx * (DD / 4) + tid];
        acc.x += a * xv.x; acc.y += a * xv.y;
        acc.z += a * xv.z; acc.w += a * xv.w;
    }
    ((float4*)(out + (size_t)BB * DD))[b * (DD / 4) + tid] = acc;       // u_att
    ((float4*)out)[b * (DD / 4) + tid] = xb[(SS - 1) * (DD / 4) + tid]; // u_last
}

// ===========================================================================
extern "C" void kernel_launch(void* const* d_in, const int* in_sizes, int n_in,
                              void* d_out, int out_size, void* d_ws, size_t ws_size,
                              hipStream_t stream) {
    const float* x    = (const float*)d_in[0];
    const int*   mask = (const int*)  d_in[1];
    const float* W1   = (const float*)d_in[2];
    const float* b1   = (const float*)d_in[3];
    const float* w2   = (const float*)d_in[4];
    float* out = (float*)d_out;

    float*          scores = (float*)d_ws;                       // 256 KB
    unsigned short* w1t    = (unsigned short*)(scores + ROWS);   // 2 MB bf16

    // allow 284,672 B of dynamic LDS (CDNA5 WGP has 320 KB)
    (void)hipFuncSetAttribute((const void*)gemm_tanh_score_kernel,
                              hipFuncAttributeMaxDynamicSharedMemorySize,
                              SMEM_BYTES);

    transpose_w1_kernel   <<<(DD / 32) * (DD / 32), 256, 0, stream>>>(W1, w1t);
    gemm_tanh_score_kernel<<<ROWS / M_TILE, 256, SMEM_BYTES, stream>>>(x, w1t, b1, w2, scores);
    softmax_attend_kernel <<<BB, 256, 0, stream>>>(x, mask, scores, out);
}
